// DSNet_8564164788859
// MI455X (gfx1250) — matrix operand order repaired and checked by reading, not compiled, not run on hardware
//
#include <hip/hip_runtime.h>
#include <hip/hip_bf16.h>

typedef __attribute__((ext_vector_type(16))) _Float16 v16h;
typedef __attribute__((ext_vector_type(8)))  _Float16 v8h;
typedef __attribute__((ext_vector_type(8)))  float    v8f;

#define T_REF   65536
#define NCHUNK  65536
#define NHALF   32768
#define CDIM    512
#define KCONV   416   // 136*3 = 408 padded to multiple of 32

__device__ __forceinline__ v8h load8h(const _Float16* p) {
    return *reinterpret_cast<const v8h*>(p);
}
__device__ __forceinline__ v16h cat16(v8h a, v8h b) {
    return __builtin_shufflevector(a, b, 0,1,2,3,4,5,6,7,8,9,10,11,12,13,14,15);
}

// ---------------------------------------------------------------------------
// Weight preprocessing: f32 -> f16, im2col W_seq with K pad, center taps of
// W2/W3/W4 (W4 padded from 136 to 144 output rows).
// ---------------------------------------------------------------------------
__global__ __launch_bounds__(256) void prep_weights(
    const float* __restrict__ Wseq, const float* __restrict__ W2,
    const float* __restrict__ W3,   const float* __restrict__ W4,
    _Float16* __restrict__ WseqH, _Float16* __restrict__ W2H,
    _Float16* __restrict__ W3H,   _Float16* __restrict__ W4H)
{
    int gid = blockIdx.x * 256 + threadIdx.x;
    if (gid < 512 * KCONV) {                       // [512][416]
        int k = gid % KCONV, co = gid / KCONV;
        WseqH[gid] = (k < 408) ? (_Float16)Wseq[co * 408 + k] : (_Float16)0.f;
    }
    if (gid < 256 * 512) W2H[gid] = (_Float16)W2[gid * 3 + 1];
    if (gid < 128 * 256) W3H[gid] = (_Float16)W3[gid * 3 + 1];
    if (gid < 144 * 128) {                         // padded N: 136 -> 144
        int n = gid / 128, k = gid % 128;
        W4H[gid] = (n < 136) ? (_Float16)W4[(n * 128 + k) * 3 + 1] : (_Float16)0.f;
    }
}

// ---------------------------------------------------------------------------
// im2col of refer: Aim[pos][k], k = cin*3 + dt, value refer[cin][pos-1+dt]
// with zero padding at sequence edges and k >= 408.
// ---------------------------------------------------------------------------
__global__ __launch_bounds__(256) void im2col_refer(
    const float* __restrict__ refer, _Float16* __restrict__ Aim)
{
    long gid = (long)blockIdx.x * 256 + threadIdx.x;
    if (gid >= (long)T_REF * KCONV) return;
    int  k   = (int)(gid % KCONV);
    long pos = gid / KCONV;
    _Float16 v = (_Float16)0.f;
    if (k < 408) {
        int cin = k / 3, dt = k % 3;
        long t = pos - 1 + dt;
        if (t >= 0 && t < T_REF) v = (_Float16)refer[(size_t)cin * T_REF + t];
    }
    Aim[gid] = v;
}

// ---------------------------------------------------------------------------
// Fused conv1d(refer) + linear1 for chunks 0..32767.
// Block = 128 threads (4 waves). Block computes a [16 pos x 512 ch] stripe of
// the conv as WMMA GEMM (each wave: 8 N-tiles, K loop 416/32), adds b_seq,
// parks it in LDS, then applies the 3-tap channel stencil + ReLU -> s0 (f16).
// (Round-1 structure: per-tile B load directly feeding its WMMA — the
// compiler fully unrolls and clause-schedules this best.)
// ---------------------------------------------------------------------------
__global__ __launch_bounds__(128) void conv_linear1_kernel(
    const _Float16* __restrict__ Aim,   // [65536][416]
    const _Float16* __restrict__ Wh,    // [512][416]
    const float* __restrict__ bseq,     // [512]
    const float* __restrict__ W1,       // [6]
    const float* __restrict__ b1,       // [1]
    _Float16* __restrict__ s0)          // [65536][512]
{
    __shared__ float sm[16][CDIM];
    const int lane = threadIdx.x & 31;
    const int wave = threadIdx.x >> 5;
    const int pos0 = blockIdx.x * 16;
    const int m    = lane & 15;
    const int hi   = lane >> 4;
    const int nbase = wave * 128;       // 8 tiles x 16 channels per wave

    v8f acc[8];
    #pragma unroll
    for (int t = 0; t < 8; ++t) acc[t] = (v8f){0.f,0.f,0.f,0.f,0.f,0.f,0.f,0.f};

    const _Float16* arow = Aim + (size_t)(pos0 + m) * KCONV;
    for (int k0 = 0; k0 < KCONV; k0 += 32) {
        v16h af = cat16(load8h(arow + k0 + hi * 8),
                        load8h(arow + k0 + 16 + hi * 8));
        #pragma unroll
        for (int t = 0; t < 8; ++t) {
            const _Float16* brow = Wh + (size_t)(nbase + t * 16 + m) * KCONV;
            v16h bf = cat16(load8h(brow + k0 + hi * 8),
                            load8h(brow + k0 + 16 + hi * 8));
            acc[t] = __builtin_amdgcn_wmma_f32_16x16x32_f16(
                false, af, false, bf, (short)0, acc[t], false, false);
        }
    }

    // Epilogue: D element r -> (M = r + 8*hi, N = lane%16); add bias, to LDS.
    #pragma unroll
    for (int t = 0; t < 8; ++t) {
        int col = nbase + t * 16 + m;
        float bv = bseq[col];
        #pragma unroll
        for (int r = 0; r < 8; ++r)
            sm[hi * 8 + r][col] = acc[t][r] + bv;
    }
    __syncthreads();

    // linear1: 8 chunks x 512 channels, 3-tap stencil over channel axis.
    const float w00 = W1[0], w01 = W1[1], w02 = W1[2];
    const float w10 = W1[3], w11 = W1[4], w12 = W1[5], bb = b1[0];
    for (int idx = threadIdx.x; idx < 8 * CDIM; idx += 128) {
        int ci = idx >> 9;
        int c  = idx & (CDIM - 1);
        const float* ar = sm[2 * ci];
        const float* br = sm[2 * ci + 1];
        float am = (c > 0)        ? ar[c - 1] : 0.f;
        float ap = (c < CDIM - 1) ? ar[c + 1] : 0.f;
        float bm = (c > 0)        ? br[c - 1] : 0.f;
        float bp = (c < CDIM - 1) ? br[c + 1] : 0.f;
        float v = w00*am + w01*ar[c] + w02*ap + w10*bm + w11*br[c] + w12*bp + bb;
        v = v > 0.f ? v : 0.f;
        s0[(size_t)(pos0 / 2 + ci) * CDIM + c] = (_Float16)v;
    }
}

// ---------------------------------------------------------------------------
// linear1 on the x-half: chunks 32768..65535 read x rows directly (f32).
// ---------------------------------------------------------------------------
__global__ __launch_bounds__(256) void linear1_x_kernel(
    const float* __restrict__ x, const float* __restrict__ W1,
    const float* __restrict__ b1, _Float16* __restrict__ s0)
{
    long gid = (long)blockIdx.x * 256 + threadIdx.x;   // NHALF*512 total
    if (gid >= (long)NHALF * CDIM) return;
    int  c  = (int)(gid & (CDIM - 1));
    long m2 = gid >> 9;
    const float* ar = x + (size_t)(2 * m2) * CDIM;
    const float* br = ar + CDIM;
    float am = (c > 0)        ? ar[c - 1] : 0.f;
    float ap = (c < CDIM - 1) ? ar[c + 1] : 0.f;
    float bm = (c > 0)        ? br[c - 1] : 0.f;
    float bp = (c < CDIM - 1) ? br[c + 1] : 0.f;
    float v = W1[0]*am + W1[1]*ar[c] + W1[2]*ap
            + W1[3]*bm + W1[4]*br[c] + W1[5]*bp + b1[0];
    v = v > 0.f ? v : 0.f;
    s0[(size_t)(NHALF + m2) * CDIM + c] = (_Float16)v;
}

// ---------------------------------------------------------------------------
// Generic WMMA GEMM: out = act(A[M][K] @ W[Npad][K]^T + bias).
// Fully compile-time inner structure: NT tiles per wave (all active), K known
// at compile time (loop fully unrollable), no guards inside loops. The only
// runtime guard is a single wave-uniform early-return at the top (used by the
// last layer where one wave has no tiles). Lane-level Nout masking happens
// only in the f32 store epilogue (after all WMMAs).
// ---------------------------------------------------------------------------
template<int NT, int K, bool RELU, bool F16OUT>
__global__ __launch_bounds__(128) void gemm_kernel(
    const _Float16* __restrict__ A, const _Float16* __restrict__ W,
    const float* __restrict__ bias,
    int Npad, int Nout,
    _Float16* __restrict__ outH, float* __restrict__ outF, int ldo)
{
    const int lane = threadIdx.x & 31;
    const int wave = threadIdx.x >> 5;
    const int m0 = blockIdx.x * 16;
    const int n0 = wave * (NT * 16);
    if (n0 >= Npad) return;                      // wave-uniform, outside loops
    const int m  = lane & 15;
    const int hi = lane >> 4;

    v8f acc[NT];
    #pragma unroll
    for (int t = 0; t < NT; ++t) acc[t] = (v8f){0.f,0.f,0.f,0.f,0.f,0.f,0.f,0.f};

    const _Float16* arow = A + (size_t)(m0 + m) * K;
    const _Float16* wbase = W + (size_t)(n0 + m) * K;
    for (int k0 = 0; k0 < K; k0 += 32) {
        v16h af = cat16(load8h(arow + k0 + hi * 8),
                        load8h(arow + k0 + 16 + hi * 8));
        #pragma unroll
        for (int t = 0; t < NT; ++t) {
            v16h bf = cat16(load8h(wbase + (size_t)t * 16 * K + k0 + hi * 8),
                            load8h(wbase + (size_t)t * 16 * K + k0 + 16 + hi * 8));
            acc[t] = __builtin_amdgcn_wmma_f32_16x16x32_f16(
                false, af, false, bf, (short)0, acc[t], false, false);
        }
    }

    #pragma unroll
    for (int t = 0; t < NT; ++t) {
        int col = n0 + t * 16 + m;
        float bv = (col < Nout) ? bias[col] : 0.f;
        #pragma unroll
        for (int r = 0; r < 8; ++r) {
            float v = acc[t][r] + bv;
            if (RELU) v = v > 0.f ? v : 0.f;
            size_t mrow = (size_t)(m0 + hi * 8 + r);
            if (F16OUT) outH[mrow * ldo + col] = (_Float16)v;
            else if (col < Nout) outF[mrow * ldo + col] = v;
        }
    }
}

// ---------------------------------------------------------------------------
extern "C" void kernel_launch(void* const* d_in, const int* in_sizes, int n_in,
                              void* d_out, int out_size, void* d_ws, size_t ws_size,
                              hipStream_t stream) {
    const float* refer = (const float*)d_in[1];
    const float* x     = (const float*)d_in[2];
    const float* Wseq  = (const float*)d_in[3];
    const float* bseq  = (const float*)d_in[4];
    const float* W1    = (const float*)d_in[5];
    const float* b1    = (const float*)d_in[6];
    const float* W2    = (const float*)d_in[7];
    const float* b2    = (const float*)d_in[8];
    const float* W3    = (const float*)d_in[9];
    const float* b3    = (const float*)d_in[10];
    const float* W4    = (const float*)d_in[11];
    const float* b4    = (const float*)d_in[12];

    size_t off = 0;
    auto take = [&](size_t bytes) {
        void* p = (char*)d_ws + off;
        off += (bytes + 255) & ~(size_t)255;
        return p;
    };
    _Float16* Aim   = (_Float16*)take((size_t)T_REF * KCONV * 2);  // 54.5 MB
    _Float16* WseqH = (_Float16*)take((size_t)512 * KCONV * 2);
    _Float16* W2H   = (_Float16*)take((size_t)256 * 512 * 2);
    _Float16* W3H   = (_Float16*)take((size_t)128 * 256 * 2);
    _Float16* W4H   = (_Float16*)take((size_t)144 * 128 * 2);
    _Float16* s0    = (_Float16*)take((size_t)NCHUNK * 512 * 2);   // 64 MB
    _Float16* s1    = (_Float16*)take((size_t)NCHUNK * 256 * 2);   // 32 MB
    _Float16* s2    = (_Float16*)take((size_t)NCHUNK * 128 * 2);   // 16 MB

    // 1) weights f32->f16 (+ im2col W_seq, padding)
    prep_weights<<<(512 * KCONV + 255) / 256, 256, 0, stream>>>(
        Wseq, W2, W3, W4, WseqH, W2H, W3H, W4H);

    // 2) im2col refer -> A matrix [65536 x 416] f16
    im2col_refer<<<(int)(((long)T_REF * KCONV + 255) / 256), 256, 0, stream>>>(
        refer, Aim);

    // 3) conv (WMMA GEMM) + fused bias + linear1 stencil -> s0[0..32767]
    conv_linear1_kernel<<<T_REF / 16, 128, 0, stream>>>(
        Aim, WseqH, bseq, W1, b1, s0);

    // 4) linear1 on x -> s0[32768..65535]
    linear1_x_kernel<<<(int)(((long)NHALF * CDIM + 255) / 256), 256, 0, stream>>>(
        x, W1, b1, s0);

    // 5) layer2: [65536x512] @ [512x256]^T + b2, ReLU -> s1
    //    4 waves x 4 tiles = 256 cols, single pass over A.
    gemm_kernel<4, 512, true, true><<<NCHUNK / 16, 128, 0, stream>>>(
        s0, W2H, b2, 256, 256, s1, nullptr, 256);

    // 6) layer3: [65536x256] @ [256x128]^T + b3, ReLU -> s2
    //    4 waves x 2 tiles = 128 cols.
    gemm_kernel<2, 256, true, true><<<NCHUNK / 16, 128, 0, stream>>>(
        s1, W3H, b3, 128, 128, s2, nullptr, 128);

    // 7) layer4: [65536x128] @ [128x144pad]^T + b4 -> d_out (f32, c<136)
    //    waves 0-2 x 3 tiles = 9 tiles (144 cols); wave 3 early-returns.
    gemm_kernel<3, 128, false, false><<<NCHUNK / 16, 128, 0, stream>>>(
        s2, W4H, b4, 144, 136, nullptr, (float*)d_out, 136);
}
